// CustomTransformerLayer_57380763074946
// MI455X (gfx1250) — compile-verified, matrix operand order checked by reading, not run on hardware
//
#include <hip/hip_runtime.h>
#include <hip/hip_bf16.h>
#include <math.h>

// ---------------------------------------------------------------------------
// Shapes (compile-time constants from the reference)
// ---------------------------------------------------------------------------
#define Bn   4
#define Sn   1024
#define Hn   1280
#define NHn  20
#define DHn  64
#define In   5120
#define Cn   13
#define Pn   8
#define BSn  (Bn * Sn)          // 4096 rows
#define LN_EPS 1e-5f

typedef _Float16 half8  __attribute__((ext_vector_type(8)));
typedef _Float16 v16h   __attribute__((ext_vector_type(16)));
typedef float    v8f    __attribute__((ext_vector_type(8)));
typedef unsigned int u32x4 __attribute__((ext_vector_type(4)));
typedef int          i32x4 __attribute__((ext_vector_type(4)));
typedef int          i32x8 __attribute__((ext_vector_type(8)));
// GCC-style vector matching the async-builtin's pointee type exactly
typedef int i32x4v __attribute__((vector_size(16)));

#define AS1 __attribute__((address_space(1)))
#define AS3 __attribute__((address_space(3)))

// --- CDNA5 async/TDM feature detection (fallback to plain staging) ---------
#if __has_builtin(__builtin_amdgcn_global_load_async_to_lds_b128)
#define HAVE_ASYNC_LDS 1
#else
#define HAVE_ASYNC_LDS 0
#endif
#if __has_builtin(__builtin_amdgcn_tensor_load_to_lds)
#define HAVE_TDM 1
#else
#define HAVE_TDM 0
#endif

__device__ __forceinline__ AS1 i32x4v* as1_v4(const void* p) {
    return (AS1 i32x4v*)(unsigned long long)p;
}
__device__ __forceinline__ AS3 i32x4v* as3_v4(const void* p) {
    // generic LDS pointers keep the LDS byte offset in the low 32 bits
    return (AS3 i32x4v*)(unsigned int)(unsigned long long)p;
}
__device__ __forceinline__ void wait_asynccnt0() {
#if __has_builtin(__builtin_amdgcn_s_wait_asynccnt)
    __builtin_amdgcn_s_wait_asynccnt(0);
#else
    asm volatile("s_wait_asynccnt 0x0" ::: "memory");
#endif
}

// ---------------------------------------------------------------------------
// f32 -> f16 conversion
// ---------------------------------------------------------------------------
__global__ void cvt_f32_to_f16(const float* __restrict__ in,
                               _Float16* __restrict__ out, int n) {
    int i = blockIdx.x * 256 + threadIdx.x;
    if (i < n) out[i] = (_Float16)in[i];
}

// ---------------------------------------------------------------------------
// Crosstalk bilinear: R = out1 @ out2   (tiny: P=8, C=13)
// ---------------------------------------------------------------------------
__global__ void crosstalk_kernel(const float* __restrict__ M,
                                 const float* __restrict__ W1, const float* __restrict__ b1,
                                 const float* __restrict__ W2, const float* __restrict__ b2,
                                 float* __restrict__ R) {
    int t = threadIdx.x;
    if (t >= Pn * Pn) return;
    int i = t / Pn, j = t % Pn;
    float acc = 0.f;
    for (int c = 0; c < Cn; ++c) {
        float o1 = b1[i];
        for (int k = 0; k < Cn; ++k) o1 += M[k * Cn + c] * W1[i * Cn + k];
        float o2 = b2[j];
        for (int k = 0; k < Cn; ++k) o2 += M[c * Cn + k] * W2[j * Cn + k];
        acc += o1 * o2;
    }
    R[i * Pn + j] = acc;
}

// ---------------------------------------------------------------------------
// ptm projection: ptm[row,p] = h[row,:] . W_ptm[p,:] + b[p]
// ---------------------------------------------------------------------------
__global__ void ptm_proj_kernel(const float* __restrict__ h,
                                const float* __restrict__ W, const float* __restrict__ bias,
                                float* __restrict__ ptm) {
    int idx = blockIdx.x * 256 + threadIdx.x;
    if (idx >= BSn * Pn) return;
    int p = idx & (Pn - 1);
    int row = idx / Pn;
    const float* hr = h + (size_t)row * Hn;
    const float* wr = W + (size_t)p * Hn;
    float acc = bias[p];
    for (int k = 0; k < Hn; ++k) acc += hr[k] * wr[k];
    ptm[idx] = acc;
}

// softmax over P=8 (in place) then t = ptm @ R ; one thread per (b,s) row
__global__ void ptm_softmax_t_kernel(float* __restrict__ ptm,
                                     const float* __restrict__ R,
                                     float* __restrict__ tR) {
    int row = blockIdx.x * 256 + threadIdx.x;
    if (row >= BSn) return;
    float v[Pn];
    float mx = -3.4e38f;
    for (int p = 0; p < Pn; ++p) { v[p] = ptm[row * Pn + p]; mx = fmaxf(mx, v[p]); }
    float s = 0.f;
    for (int p = 0; p < Pn; ++p) { v[p] = __expf(v[p] - mx); s += v[p]; }
    float inv = 1.f / s;
    for (int p = 0; p < Pn; ++p) { v[p] *= inv; ptm[row * Pn + p] = v[p]; }
    for (int d = 0; d < Pn; ++d) {
        float acc = 0.f;
        for (int c = 0; c < Pn; ++c) acc += v[c] * R[c * Pn + d];
        tR[row * Pn + d] = acc;
    }
}

// Bb[b,l,m] = tanh( t[b,l,:] . ptm[b,m,:] ) * scale
__global__ void bias_mat_kernel(const float* __restrict__ tR,
                                const float* __restrict__ ptm,
                                const float* __restrict__ scale,
                                float* __restrict__ Bb) {
    size_t idx = (size_t)blockIdx.x * 256 + threadIdx.x;
    if (idx >= (size_t)Bn * Sn * Sn) return;
    int m = (int)(idx % Sn);
    int l = (int)((idx / Sn) % Sn);
    int b = (int)(idx / ((size_t)Sn * Sn));
    const float* tl = tR  + ((size_t)b * Sn + l) * Pn;
    const float* pm = ptm + ((size_t)b * Sn + m) * Pn;
    float acc = 0.f;
#pragma unroll
    for (int p = 0; p < Pn; ++p) acc += tl[p] * pm[p];
    Bb[idx] = tanhf(acc) * scale[0];
}

// ---------------------------------------------------------------------------
// Tiled WMMA GEMM:  out[M,N] = A[M,K](f16) @ W[N,K]^T(f16) + bias
// Block tile 128x64, BK=32, 256 threads = 8 waves, each wave a 32x32 patch.
// A tile staged with GLOBAL_LOAD_ASYNC_TO_LDS (ASYNCcnt) when available;
// W tile staged with the Tensor Data Mover (TENSORcnt) when available.
// epilogue: 0 = store f16 ; 1 = exact GELU -> f16 ; 2 = residual add -> f32
// ---------------------------------------------------------------------------
#define GBM 128
#define GBN 64
#define GBK 32

__global__ __launch_bounds__(256) void gemm_wmma_kernel(
    const _Float16* __restrict__ A, const _Float16* __restrict__ W,
    const float* __restrict__ bias, int M, int N, int K, int epilogue,
    _Float16* __restrict__ out16, float* __restrict__ out32,
    const float* __restrict__ resid) {
    __shared__ _Float16 sA[GBM * GBK];
    __shared__ _Float16 sB[GBN * GBK];

    const int tid  = threadIdx.x;
    const int lane = tid & 31;
    const int wave = tid >> 5;        // 0..7
    const int wm   = wave >> 1;       // 0..3 row group (32 rows each)
    const int wn   = wave & 1;        // 0..1 col group (32 cols each)
    const int bm   = blockIdx.y * GBM;
    const int bn   = blockIdx.x * GBN;

    const int mrow = lane & 15;
    const int hi2  = lane >> 4;
    const int kb2  = hi2 << 3;        // 0 or 8 (K sub-base per half-wave)

    v8f c[2][2];
    for (int i = 0; i < 2; ++i)
        for (int j = 0; j < 2; ++j) { v8f z = {}; c[i][j] = z; }

    for (int k0 = 0; k0 < K; k0 += GBK) {
        // ---- Stage A tile: 128 rows x 32 halves (8 KB), 2 threads per row
        {
            int r  = tid >> 1;
            int c8 = (tid & 1) * 16;
            const _Float16* src = A + (size_t)(bm + r) * K + k0 + c8;
            _Float16* dst = sA + r * GBK + c8;
#if HAVE_ASYNC_LDS
            __builtin_amdgcn_global_load_async_to_lds_b128(as1_v4(src), as3_v4(dst), 0, 0);
            __builtin_amdgcn_global_load_async_to_lds_b128(as1_v4(src), as3_v4(dst), 16, 0);
#else
            *(half8*)(dst)     = *(const half8*)(src);
            *(half8*)(dst + 8) = *(const half8*)(src + 8);
#endif
            if (k0 + GBK < K) __builtin_prefetch(src + GBK, 0, 1);
        }
        // ---- Stage W tile: 64 rows x 32 halves (4 KB)
#if HAVE_TDM
        {
            // D# for a 2-D f16 tensor (N x K) , tile GBN x GBK at (bn, k0)
            unsigned long long gaddr =
                (unsigned long long)(const void*)(W + (size_t)bn * K + k0);
            unsigned int ldsb = (unsigned int)(unsigned long long)(const void*)sB;
            u32x4 g0;
            g0[0] = 1u;                                        // count=1 user D#
            g0[1] = ldsb;                                      // lds_addr
            g0[2] = (unsigned int)(gaddr & 0xffffffffu);       // global_addr[31:0]
            g0[3] = (unsigned int)((gaddr >> 32) & 0x01ffffffu) | (2u << 30); // addr[56:32]|type=2
            i32x8 g1;
            g1[0] = (1 << 16);                                 // data_size = 2 bytes
            g1[1] = (int)(((unsigned)K & 0xffffu) << 16);      // tensor_dim0 lo16
            g1[2] = (int)((((unsigned)K >> 16) & 0xffffu) |
                          (((unsigned)N & 0xffffu) << 16));    // dim0 hi16 | dim1 lo16
            g1[3] = (int)((((unsigned)N >> 16) & 0xffffu) |
                          ((unsigned)GBK << 16));              // dim1 hi16 | tile_dim0
            g1[4] = GBN;                                       // tile_dim1 ; tile_dim2=0
            g1[5] = K;                                         // tensor_dim0_stride[31:0]
            g1[6] = 0;
            g1[7] = 0;
            i32x4 gz; gz[0] = 0; gz[1] = 0; gz[2] = 0; gz[3] = 0;
            if (wave == 0) {
#if __clang_major__ >= 23
                i32x8 gz8;
                for (int q = 0; q < 8; ++q) gz8[q] = 0;
                __builtin_amdgcn_tensor_load_to_lds(g0, g1, gz, gz, gz8, 0);
#else
                __builtin_amdgcn_tensor_load_to_lds(g0, g1, gz, gz, 0);
#endif
            }
        }
#else
        if (tid < 128) {
            int r  = tid >> 1;
            int c8 = (tid & 1) * 16;
            const _Float16* src = W + (size_t)(bn + r) * K + k0 + c8;
            *(half8*)(sB + r * GBK + c8)     = *(const half8*)(src);
            *(half8*)(sB + r * GBK + c8 + 8) = *(const half8*)(src + 8);
        }
#endif
        // ---- make staged data visible to all waves
#if HAVE_ASYNC_LDS
        wait_asynccnt0();
#endif
#if HAVE_TDM
        __builtin_amdgcn_s_wait_tensorcnt(0);
#endif
        __syncthreads();

#pragma unroll
        for (int i = 0; i < 2; ++i) {
            const int ar = wm * 32 + i * 16 + mrow;    // A-fragment row (M = lane&15)
            v16h a;
            {
                half8 lo = *(const half8*)(sA + ar * GBK + kb2);
                half8 hi = *(const half8*)(sA + ar * GBK + 16 + kb2);
#pragma unroll
                for (int e = 0; e < 8; ++e) { a[e] = lo[e]; a[e + 8] = hi[e]; }
            }
#pragma unroll
            for (int j = 0; j < 2; ++j) {
                const int br = wn * 32 + j * 16 + mrow; // B-fragment column (N = lane&15)
                v16h bf;
                {
                    half8 lo = *(const half8*)(sB + br * GBK + kb2);
                    half8 hi = *(const half8*)(sB + br * GBK + 16 + kb2);
#pragma unroll
                    for (int e = 0; e < 8; ++e) { bf[e] = lo[e]; bf[e + 8] = hi[e]; }
                }
                c[i][j] = __builtin_amdgcn_wmma_f32_16x16x32_f16(
                    false, a, false, bf, (short)0, c[i][j], false, false);
            }
        }
        __syncthreads();
    }

    // Epilogue. C layout: VGPR r -> row = r + 8*hi2 ; col = lane&15.
#pragma unroll
    for (int i = 0; i < 2; ++i)
#pragma unroll
        for (int j = 0; j < 2; ++j) {
            const int gcol = bn + wn * 32 + j * 16 + mrow;
            const float bv = bias ? bias[gcol] : 0.f;
#pragma unroll
            for (int r = 0; r < 8; ++r) {
                const int grow = bm + wm * 32 + i * 16 + r + 8 * hi2;
                float val = c[i][j][r] + bv;
                const size_t idx = (size_t)grow * N + gcol;
                if (epilogue == 1) {
                    val = 0.5f * val * (1.f + erff(val * 0.70710678f));
                    out16[idx] = (_Float16)val;
                } else if (epilogue == 0) {
                    out16[idx] = (_Float16)val;
                } else {
                    out32[idx] = resid[idx] + val;
                }
            }
        }
}

// ---------------------------------------------------------------------------
// WMMA attention: one wave per (b, head, 16-query-row tile).
// 16 x S score strip in LDS (64 KB), full-row softmax, WMMA for QK^T and PV.
// ---------------------------------------------------------------------------
__global__ __launch_bounds__(32) void attn_wmma_kernel(
    const _Float16* __restrict__ q16, const _Float16* __restrict__ k16,
    const _Float16* __restrict__ v16, const float* __restrict__ Bb,
    const float* __restrict__ mask, float* __restrict__ ctx) {
    __shared__ float sc[16 * Sn];     // 64 KB score strip
    __shared__ float rinv[16];

    const int lane = threadIdx.x;
    const int blk  = blockIdx.x;
    const int qt   = blk & (Sn / 16 - 1);          // 0..63
    const int h    = (blk >> 6) % NHn;
    const int b    = blk / ((Sn / 16) * NHn);
    const int mrow = lane & 15;
    const int hi2  = lane >> 4;
    const int kb2  = hi2 << 3;
    const int qbase = qt * 16;

    // Q fragments (16 x 64, two K=32 fragments)
    v16h qa0, qa1;
    {
        const _Float16* qp = q16 + ((size_t)(b * Sn + qbase + mrow)) * Hn + h * DHn;
        half8 l0 = *(const half8*)(qp + kb2);
        half8 h0 = *(const half8*)(qp + 16 + kb2);
        half8 l1 = *(const half8*)(qp + 32 + kb2);
        half8 h1 = *(const half8*)(qp + 48 + kb2);
#pragma unroll
        for (int e = 0; e < 8; ++e) {
            qa0[e] = l0[e]; qa0[e + 8] = h0[e];
            qa1[e] = l1[e]; qa1[e + 8] = h1[e];
        }
    }

    // ---- scores: 64 key tiles of 16 -------------------------------------
    for (int kt = 0; kt < Sn / 16; ++kt) {
        v16h ka0, ka1;
        {
            const _Float16* kp = k16 + ((size_t)(b * Sn + kt * 16 + mrow)) * Hn + h * DHn;
            half8 l0 = *(const half8*)(kp + kb2);
            half8 h0 = *(const half8*)(kp + 16 + kb2);
            half8 l1 = *(const half8*)(kp + 32 + kb2);
            half8 h1 = *(const half8*)(kp + 48 + kb2);
#pragma unroll
            for (int e = 0; e < 8; ++e) {
                ka0[e] = l0[e]; ka0[e + 8] = h0[e];
                ka1[e] = l1[e]; ka1[e + 8] = h1[e];
            }
        }
        v8f cc = {};
        cc = __builtin_amdgcn_wmma_f32_16x16x32_f16(false, qa0, false, ka0,
                                                    (short)0, cc, false, false);
        cc = __builtin_amdgcn_wmma_f32_16x16x32_f16(false, qa1, false, ka1,
                                                    (short)0, cc, false, false);
#pragma unroll
        for (int r = 0; r < 8; ++r) {
            const int row = r + 8 * hi2;
            const int col = kt * 16 + mrow;
            float s = cc[r] * 0.125f;                         // 1/sqrt(64)
            s += Bb[((size_t)b * Sn + qbase + row) * Sn + col];
            s += (1.f - mask[b * Sn + col]) * -3.4028235e38f;
            sc[row * Sn + col] = s;
        }
    }
    __syncthreads();

    // ---- softmax across S per row ---------------------------------------
    for (int r = 0; r < 16; ++r) {
        float mx = -3.4e38f;
        for (int c2 = lane; c2 < Sn; c2 += 32) mx = fmaxf(mx, sc[r * Sn + c2]);
        for (int off = 16; off; off >>= 1) mx = fmaxf(mx, __shfl_xor(mx, off, 32));
        float sum = 0.f;
        for (int c2 = lane; c2 < Sn; c2 += 32) {
            float e = __expf(sc[r * Sn + c2] - mx);
            sc[r * Sn + c2] = e;
            sum += e;
        }
        for (int off = 16; off; off >>= 1) sum += __shfl_xor(sum, off, 32);
        if (lane == 0) rinv[r] = 1.f / sum;
    }
    __syncthreads();

    // ---- PV: A = probs (16 x S), B = V (S x 64) -------------------------
    v8f co[4];
    for (int nt = 0; nt < 4; ++nt) { v8f z = {}; co[nt] = z; }
    const float pinv = rinv[mrow];
    for (int kt = 0; kt < Sn / 32; ++kt) {
        v16h pa;
        {
            const float* pr = sc + mrow * Sn + kt * 32;
#pragma unroll
            for (int e = 0; e < 16; ++e) {
                const int kk = (e < 8) ? (kb2 + e) : (16 + kb2 + (e - 8));
                pa[e] = (_Float16)(pr[kk] * pinv);
            }
        }
#pragma unroll
        for (int nt = 0; nt < 4; ++nt) {
            v16h vb;
#pragma unroll
            for (int e = 0; e < 16; ++e) {
                const int kk = (e < 8) ? (kb2 + e) : (16 + kb2 + (e - 8));
                vb[e] = v16[((size_t)(b * Sn + kt * 32 + kk)) * Hn + h * DHn + nt * 16 + mrow];
            }
            co[nt] = __builtin_amdgcn_wmma_f32_16x16x32_f16(false, pa, false, vb,
                                                            (short)0, co[nt], false, false);
        }
    }
#pragma unroll
    for (int nt = 0; nt < 4; ++nt)
#pragma unroll
        for (int r = 0; r < 8; ++r) {
            const int row = qbase + r + 8 * hi2;
            const int col = h * DHn + nt * 16 + mrow;
            ctx[((size_t)b * Sn + row) * Hn + col] = co[nt][r];
        }
}

// ---------------------------------------------------------------------------
// residual + LayerNorm ; emits f32 x (for final residual) and f16 x16 (FFN in)
// ---------------------------------------------------------------------------
__global__ __launch_bounds__(256) void residual_ln_kernel(
    const float* __restrict__ hsrc, const float* __restrict__ ctx,
    const float* __restrict__ g, const float* __restrict__ be,
    float* __restrict__ x, _Float16* __restrict__ x16) {
    const int row = blockIdx.x;
    const int tid = threadIdx.x;
    const int lane = tid & 31;
    const int wave = tid >> 5;
    const size_t base = (size_t)row * Hn;

    float vv[5];
    float s1 = 0.f, s2 = 0.f;
    int n = 0;
    for (int k = tid; k < Hn; k += 256, ++n) {
        float v = hsrc[base + k] + ctx[base + k];
        vv[n] = v; s1 += v; s2 += v * v;
    }
    for (int off = 16; off; off >>= 1) {
        s1 += __shfl_xor(s1, off, 32);
        s2 += __shfl_xor(s2, off, 32);
    }
    __shared__ float a1[8], a2[8];
    if (lane == 0) { a1[wave] = s1; a2[wave] = s2; }
    __syncthreads();
    float t1 = 0.f, t2 = 0.f;
    for (int w = 0; w < 8; ++w) { t1 += a1[w]; t2 += a2[w]; }
    const float mu = t1 / (float)Hn;
    const float var = t2 / (float)Hn - mu * mu;
    const float rstd = rsqrtf(var + LN_EPS);
    n = 0;
    for (int k = tid; k < Hn; k += 256, ++n) {
        float v = (vv[n] - mu) * rstd * g[k] + be[k];
        x[base + k] = v;
        x16[base + k] = (_Float16)v;
    }
}

// ---------------------------------------------------------------------------
// Host-side launcher
// ---------------------------------------------------------------------------
static inline size_t alignUp256(size_t x) { return (x + 255) & ~(size_t)255; }

extern "C" void kernel_launch(void* const* d_in, const int* in_sizes, int n_in,
                              void* d_out, int out_size, void* d_ws, size_t ws_size,
                              hipStream_t stream) {
    (void)in_sizes; (void)n_in; (void)out_size; (void)ws_size;
    const float* h      = (const float*)d_in[0];
    const float* amask  = (const float*)d_in[1];
    const float* Wq     = (const float*)d_in[2];
    const float* bq     = (const float*)d_in[3];
    const float* Wk     = (const float*)d_in[4];
    const float* bk     = (const float*)d_in[5];
    const float* Wv     = (const float*)d_in[6];
    const float* bv     = (const float*)d_in[7];
    const float* Wct1   = (const float*)d_in[8];
    const float* bct1   = (const float*)d_in[9];
    const float* Wct2   = (const float*)d_in[10];
    const float* bct2   = (const float*)d_in[11];
    const float* Mct    = (const float*)d_in[12];
    const float* bscale = (const float*)d_in[13];
    const float* Wptm   = (const float*)d_in[14];
    const float* bptm   = (const float*)d_in[15];
    const float* ln_g   = (const float*)d_in[16];
    const float* ln_b   = (const float*)d_in[17];
    const float* Wf1    = (const float*)d_in[18];
    const float* bf1    = (const float*)d_in[19];
    const float* Wf2    = (const float*)d_in[20];
    const float* bf2    = (const float*)d_in[21];
    float* out = (float*)d_out;

    // workspace carve-up
    char* w = (char*)d_ws;
    size_t off = 0;
    auto alloc = [&](size_t bytes) -> void* {
        void* p = w + off;
        off = alignUp256(off + bytes);
        return p;
    };
    _Float16* h16  = (_Float16*)alloc((size_t)BSn * Hn * 2);
    _Float16* qw16 = (_Float16*)alloc((size_t)Hn * Hn * 2);
    _Float16* kw16 = (_Float16*)alloc((size_t)Hn * Hn * 2);
    _Float16* vw16 = (_Float16*)alloc((size_t)Hn * Hn * 2);
    _Float16* f1w16 = (_Float16*)alloc((size_t)In * Hn * 2);
    _Float16* f2w16 = (_Float16*)alloc((size_t)Hn * In * 2);
    _Float16* q16  = (_Float16*)alloc((size_t)BSn * Hn * 2);
    _Float16* k16  = (_Float16*)alloc((size_t)BSn * Hn * 2);
    _Float16* v16  = (_Float16*)alloc((size_t)BSn * Hn * 2);
    float*    ptm  = (float*)alloc((size_t)BSn * Pn * 4);
    float*    tR   = (float*)alloc((size_t)BSn * Pn * 4);
    float*    Rm   = (float*)alloc((size_t)Pn * Pn * 4);
    float*    Bb   = (float*)alloc((size_t)Bn * Sn * Sn * 4);
    float*    ctx  = (float*)alloc((size_t)BSn * Hn * 4);
    float*    x    = (float*)alloc((size_t)BSn * Hn * 4);
    _Float16* x16  = (_Float16*)alloc((size_t)BSn * Hn * 2);
    _Float16* y16  = (_Float16*)alloc((size_t)BSn * In * 2);

    // 1) tiny crosstalk bilinear
    crosstalk_kernel<<<1, 64, 0, stream>>>(Mct, Wct1, bct1, Wct2, bct2, Rm);

    // 2) f32 -> f16 conversions
    auto cvt = [&](const float* src, _Float16* dst, int n) {
        cvt_f32_to_f16<<<(n + 255) / 256, 256, 0, stream>>>(src, dst, n);
    };
    cvt(h,   h16,  BSn * Hn);
    cvt(Wq,  qw16, Hn * Hn);
    cvt(Wk,  kw16, Hn * Hn);
    cvt(Wv,  vw16, Hn * Hn);
    cvt(Wf1, f1w16, In * Hn);
    cvt(Wf2, f2w16, Hn * In);

    // 3) QKV projections (WMMA, store f16)
    dim3 gQKV(Hn / GBN, BSn / GBM);
    gemm_wmma_kernel<<<gQKV, 256, 0, stream>>>(h16, qw16, bq, BSn, Hn, Hn, 0,
                                               q16, nullptr, nullptr);
    gemm_wmma_kernel<<<gQKV, 256, 0, stream>>>(h16, kw16, bk, BSn, Hn, Hn, 0,
                                               k16, nullptr, nullptr);
    gemm_wmma_kernel<<<gQKV, 256, 0, stream>>>(h16, vw16, bv, BSn, Hn, Hn, 0,
                                               v16, nullptr, nullptr);

    // 4) ptm predictor + softmax + t = ptm @ R
    ptm_proj_kernel<<<(BSn * Pn + 255) / 256, 256, 0, stream>>>(h, Wptm, bptm, ptm);
    ptm_softmax_t_kernel<<<(BSn + 255) / 256, 256, 0, stream>>>(ptm, Rm, tR);

    // 5) attention bias matrix Bb (B,S,S)
    {
        size_t nBb = (size_t)Bn * Sn * Sn;
        bias_mat_kernel<<<(unsigned)((nBb + 255) / 256), 256, 0, stream>>>(tR, ptm,
                                                                           bscale, Bb);
    }

    // 6) WMMA attention -> ctx
    attn_wmma_kernel<<<Bn * NHn * (Sn / 16), 32, 0, stream>>>(q16, k16, v16,
                                                              Bb, amask, ctx);

    // 7) residual + LayerNorm -> x (f32), x16 (f16)
    residual_ln_kernel<<<BSn, 256, 0, stream>>>(h, ctx, ln_g, ln_b, x, x16);

    // 8) FFN1: GELU(x @ Wf1^T + bf1) -> y16
    dim3 gF1(In / GBN, BSn / GBM);
    gemm_wmma_kernel<<<gF1, 256, 0, stream>>>(x16, f1w16, bf1, BSn, In, Hn, 1,
                                              y16, nullptr, nullptr);

    // 9) FFN2: x + (y @ Wf2^T + bf2) -> out (f32)
    dim3 gF2(Hn / GBN, BSn / GBM);
    gemm_wmma_kernel<<<gF2, 256, 0, stream>>>(y16, f2w16, bf2, BSn, Hn, In, 2,
                                              nullptr, out, x);
}
